// ETCModel_85693187489881
// MI455X (gfx1250) — matrix operand order repaired and checked by reading, not compiled, not run on hardware
//
#include <hip/hip_runtime.h>
#include <hip/hip_bf16.h>
#include <math.h>

// ---------------------------------------------------------------------------
// ETC model (2-branch cosine-sim attention w/ head-LayerNorm) for gfx1250.
// B=16, L=512, H=12, d=64, D=768, 3D=2304, C=20.
// GEMM and attention matmuls via V_WMMA_F32_16X16X32_F16 (wave32); out-proj uses scalar dot products.
// Workspace (~126 MB): xg[8192,768] (token embeds / padded wv) |
//   hbuf[8192,2304] | obuf[8192,768] | colsum[16,512] | wbuf[16,512]  (f32)
// ---------------------------------------------------------------------------

#define BB 16
#define LL 512
#define HH 12
#define DD 768
#define D3 2304
#define CC 20
#define KH 320   // hidden-branch K (300) zero-padded to a multiple of 32

typedef __attribute__((ext_vector_type(16))) _Float16 v16h;
typedef __attribute__((ext_vector_type(4)))  _Float16 v4h;
typedef __attribute__((ext_vector_type(8)))  float    v8f;

__device__ __forceinline__ v8f wmma_f16(const v16h a, const v16h b, const v8f c) {
  return __builtin_amdgcn_wmma_f32_16x16x32_f16(false, a, false, b, (short)0, c, false, false);
}

__device__ __forceinline__ float dot4(const float4 v) {
  return v.x * v.x + v.y * v.y + v.z * v.z + v.w * v.w;
}
// pack 8 scaled floats into dst[base..base+7]
__device__ __forceinline__ void pack8(v16h& dst, int base, const float4 lo,
                                      const float4 hi, float sc) {
  dst[base + 0] = (_Float16)(lo.x * sc); dst[base + 1] = (_Float16)(lo.y * sc);
  dst[base + 2] = (_Float16)(lo.z * sc); dst[base + 3] = (_Float16)(lo.w * sc);
  dst[base + 4] = (_Float16)(hi.x * sc); dst[base + 5] = (_Float16)(hi.y * sc);
  dst[base + 6] = (_Float16)(hi.z * sc); dst[base + 7] = (_Float16)(hi.w * sc);
}

// butterfly reductions inside a 16-lane half-wave (C-tile rows live in halves)
__device__ __forceinline__ float half_max(float x) {
  x = fmaxf(x, __shfl_xor(x, 1));
  x = fmaxf(x, __shfl_xor(x, 2));
  x = fmaxf(x, __shfl_xor(x, 4));
  x = fmaxf(x, __shfl_xor(x, 8));
  return x;
}
__device__ __forceinline__ float half_sum(float x) {
  x += __shfl_xor(x, 1); x += __shfl_xor(x, 2);
  x += __shfl_xor(x, 4); x += __shfl_xor(x, 8);
  return x;
}

// A-fragment (16x32 f16) from row-major LDS tile: one 32B run per lane.
__device__ __forceinline__ v16h load_frag_a(const _Float16* __restrict__ s, int ld, int lane) {
  const int m  = lane & 15;
  const int kh = (lane >> 4) * 16;
  v16h a;
#pragma unroll
  for (int t = 0; t < 8; ++t) {
    a[t]     = s[m * ld + kh + t];        // d [kh .. kh+7]
    a[t + 8] = s[m * ld + kh + 8 + t];    // d [kh+8 .. kh+15]
  }
  return a;
}

// B-fragment (32x16) from an N-major tile s[n][k]: 16 contiguous halves/lane.
__device__ __forceinline__ v16h load_frag_b_nmajor(const _Float16* __restrict__ s,
                                                   int ld, int lane) {
  const int n  = lane & 15;
  const int kb = (lane >> 4) * 16;
  v16h b;
#pragma unroll
  for (int t = 0; t < 16; ++t) b[t] = s[n * ld + kb + t];
  return b;
}

// ------------------------- token embedding gather --------------------------
__global__ __launch_bounds__(256) void gather_leaky_kernel(
    const int* __restrict__ ids, const float* __restrict__ emb,
    float* __restrict__ xg) {
  const size_t total = (size_t)BB * LL * DD;
  const size_t i4 = ((size_t)blockIdx.x * 256 + threadIdx.x) * 4;
  if (i4 >= total) return;
  const size_t row = i4 / DD, k = i4 % DD;  // DD % 4 == 0 -> one row per vec
  float4 x = *(const float4*)(emb + (size_t)ids[row] * DD + k);
  x.x = (x.x >= 0.f) ? x.x : 0.01f * x.x;
  x.y = (x.y >= 0.f) ? x.y : 0.01f * x.y;
  x.z = (x.z >= 0.f) ? x.z : 0.01f * x.z;
  x.w = (x.w >= 0.f) ? x.w : 0.01f * x.w;
  *(float4*)(xg + i4) = x;
}

// --------- zero-pad wv [8192,300] -> [8192,320] (K multiple of 32) ---------
__global__ __launch_bounds__(256) void pad_wv_kernel(
    const float* __restrict__ wv, float* __restrict__ wvp) {
  const size_t i4 = ((size_t)blockIdx.x * 256 + threadIdx.x) * 4;
  if (i4 >= (size_t)BB * LL * KH) return;
  const size_t row = i4 / KH, c = i4 % KH;   // 300 % 4 == 0: vec never straddles
  float4 x = {0.f, 0.f, 0.f, 0.f};
  if (c < 300) x = *(const float4*)(wv + row * 300 + c);
  *(float4*)(wvp + i4) = x;
}

// ------------------ h = leaky(X @ W^T + b), WMMA-tiled GEMM ----------------
// K must be a multiple of 32. grid: (N/32, M/64), block: 256 (8 waves).
__global__ __launch_bounds__(256) void gemm_bias_leaky_kernel(
    const float* __restrict__ X, const float* __restrict__ W,
    const float* __restrict__ bias, float* __restrict__ Hout,
    int M, int N, int K) {
  __shared__ _Float16 sA[64 * 32];   // row-major [m][k]
  __shared__ _Float16 sB[32 * 32];   // N-major   [n][k] (= raw W tile)
  const int tid  = threadIdx.x;
  const int lane = tid & 31;
  const int w    = tid >> 5;
  const int mi   = w >> 1, ni = w & 1;
  const int m0   = blockIdx.y * 64, n0 = blockIdx.x * 32;

  // static staging map: A -> 8 contiguous floats/thread, B -> 4/thread
  const int ar = tid >> 2, ac = (tid & 3) * 8;   // 64 rows x 32 cols
  const int br = tid >> 3, bc = (tid & 7) * 4;   // 32 rows x 32 cols
  const float* Xp = X + (size_t)(m0 + ar) * K + ac;
  const float* Wp = W + (size_t)(n0 + br) * K + bc;
  _Float16* sAp = sA + ar * 32 + ac;
  _Float16* sBp = sB + br * 32 + bc;

  v8f acc = {};
  for (int k0 = 0; k0 < K; k0 += 32) {
    const float4 x0 = *(const float4*)(Xp + k0);
    const float4 x1 = *(const float4*)(Xp + k0 + 4);
    const float4 w0 = *(const float4*)(Wp + k0);
    if (k0 + 32 < K)
      __builtin_prefetch(Xp + k0 + 32, 0, 1);  // global_prefetch_b8
    v4h a0; a0[0] = (_Float16)x0.x; a0[1] = (_Float16)x0.y;
            a0[2] = (_Float16)x0.z; a0[3] = (_Float16)x0.w;
    v4h a1; a1[0] = (_Float16)x1.x; a1[1] = (_Float16)x1.y;
            a1[2] = (_Float16)x1.z; a1[3] = (_Float16)x1.w;
    v4h b0; b0[0] = (_Float16)w0.x; b0[1] = (_Float16)w0.y;
            b0[2] = (_Float16)w0.z; b0[3] = (_Float16)w0.w;
    *(v4h*)(sAp)     = a0;
    *(v4h*)(sAp + 4) = a1;
    *(v4h*)(sBp)     = b0;
    __syncthreads();
    const v16h a   = load_frag_a(sA + mi * 16 * 32, 32, lane);
    const v16h bfr = load_frag_b_nmajor(sB + (ni * 16) * 32, 32, lane);
    acc = wmma_f16(a, bfr, acc);
    __syncthreads();
  }
  const int n = lane & 15, mb = (lane >> 4) * 8;
  const int gn = n0 + ni * 16 + n;
  const float bv = bias[gn];
#pragma unroll
  for (int v = 0; v < 8; ++v) {
    float x = acc[v] + bv;
    x = (x >= 0.f) ? x : 0.01f * x;
    Hout[(size_t)(m0 + mi * 16 + mb + v) * N + gn] = x;
  }
}

// ----- score tile: S = Kn(i-tile) @ Qn(j-tile)^T, LayerNorm over heads -----
// Each lane loads only the 32 Q floats its B-fragments need (halves are
// disjoint); row norm completed with one cross-half shuffle.
__device__ __forceinline__ void score_tile(
    const float* __restrict__ hb, int j0, int head, int lane, int tid,
    const v16h a0, const v16h a1, float g, float be,
    float* __restrict__ sLN, float* __restrict__ sMu, float* __restrict__ sRv,
    float z[8]) {
  const int n   = lane & 15;
  const int off = (lane >> 4) * 8;   // kh
  const float* qrow = hb + (size_t)(j0 + n) * D3 + head * 64;
  const float4 q0 = *(const float4*)(qrow + off);
  const float4 q1 = *(const float4*)(qrow + off + 4);
  const float4 q2 = *(const float4*)(qrow + 16 + off);
  const float4 q3 = *(const float4*)(qrow + 16 + off + 4);   // d[off..off+7], d[16+off..16+off+7]
  const float4 q4 = *(const float4*)(qrow + 32 + off);
  const float4 q5 = *(const float4*)(qrow + 32 + off + 4);
  const float4 q6 = *(const float4*)(qrow + 48 + off);
  const float4 q7 = *(const float4*)(qrow + 48 + off + 4);
  float ss = dot4(q0) + dot4(q1) + dot4(q2) + dot4(q3) +
             dot4(q4) + dot4(q5) + dot4(q6) + dot4(q7);
  ss += __shfl_xor(ss, 16);          // full 64-d row norm
  const float inv = 1.0f / fmaxf(sqrtf(ss), 1e-8f);
  v16h b0, b1;
  pack8(b0, 0, q0, q1, inv); pack8(b0, 8, q2, q3, inv);
  pack8(b1, 0, q4, q5, inv); pack8(b1, 8, q6, q7, inv);
  v8f s = {};
  s = wmma_f16(a0, b0, s);
  s = wmma_f16(a1, b1, s);
  const int mb = (lane >> 4) * 8;
  float* myLN = sLN + head * 256;
#pragma unroll
  for (int v = 0; v < 8; ++v) myLN[(mb + v) * 16 + n] = s[v];
  __syncthreads();
  // one thread per (i,j) element computes the cross-head stats exactly once
  if (tid < 256) {
    float sum = 0.f, sq = 0.f;
#pragma unroll
    for (int hh = 0; hh < HH; ++hh) {
      const float x = sLN[hh * 256 + tid];
      sum += x; sq += x * x;
    }
    const float mu = sum * (1.0f / HH);
    sMu[tid] = mu;
    sRv[tid] = rsqrtf(sq * (1.0f / HH) - mu * mu + 1e-5f);
  }
  __syncthreads();
#pragma unroll
  for (int v = 0; v < 8; ++v) {
    const int idx = (mb + v) * 16 + n;
    z[v] = (s[v] - sMu[idx]) * sRv[idx] * g + be;
  }
  __syncthreads();
}

// ---------------- fused attention: one block per (b, 16-row i-tile) --------
__global__ __launch_bounds__(384) void etc_attn_kernel(
    const float* __restrict__ hbuf, const float* __restrict__ gamma,
    const float* __restrict__ beta, float* __restrict__ obuf,
    float* __restrict__ colsum) {
  __shared__ float     sLN[HH * 256];     // 12 KB: per-head 16x16 scores
  __shared__ float     sMu[256];          //  1 KB: shared LN mean
  __shared__ float     sRv[256];          //  1 KB: shared LN rsqrt(var+eps)
  __shared__ _Float16  sP[HH * 16 * 32];  // 12 KB: per-head P transpose buf
  __shared__ _Float16  sV[32 * DD];       // 48 KB: V rows j0..j0+31, all heads
  __shared__ float     colAcc[LL];        //  2 KB: per-block column sums

  const int tid  = threadIdx.x;
  const int lane = tid & 31;
  const int head = tid >> 5;              // 0..11
  const int b    = blockIdx.y;
  const int i0   = blockIdx.x * 16;
  const float* hb = hbuf + (size_t)b * LL * D3;

  for (int i = tid; i < LL; i += 384) colAcc[i] = 0.f;

  // A fragments = normalized K rows of the i-tile; per-lane disjoint halves
  const int m    = lane & 15;
  const int offa = (lane >> 4) * 8;  // kh
  const float* krow = hb + (size_t)(i0 + m) * D3 + DD + head * 64;
  const float4 c0 = *(const float4*)(krow + offa);
  const float4 c1 = *(const float4*)(krow + offa + 4);        // d[kh..kh+7]
  const float4 c2 = *(const float4*)(krow + 16 + offa);
  const float4 c3 = *(const float4*)(krow + 16 + offa + 4);   // d[16+kh..]
  const float4 c4 = *(const float4*)(krow + 32 + offa);
  const float4 c5 = *(const float4*)(krow + 32 + offa + 4);
  const float4 c6 = *(const float4*)(krow + 48 + offa);
  const float4 c7 = *(const float4*)(krow + 48 + offa + 4);
  float ssk = dot4(c0) + dot4(c1) + dot4(c2) + dot4(c3) +
              dot4(c4) + dot4(c5) + dot4(c6) + dot4(c7);
  ssk += __shfl_xor(ssk, 16);
  const float invk = 1.0f / fmaxf(sqrtf(ssk), 1e-8f);
  v16h a0, a1;
  pack8(a0, 0, c0, c1, invk); pack8(a0, 8, c2, c3, invk);
  pack8(a1, 0, c4, c5, invk); pack8(a1, 8, c6, c7, invk);
  __syncthreads();

  const float g = gamma[head], be = beta[head];
  const int n  = lane & 15;
  const int mb = (lane >> 4) * 8;

  float m_run[8], s_run[8];
#pragma unroll
  for (int v = 0; v < 8; ++v) { m_run[v] = -INFINITY; s_run[v] = 0.f; }

  // ---- pass 1: online softmax stats over j ----
  for (int j0 = 0; j0 < LL; j0 += 16) {
    float z[8];
    score_tile(hb, j0, head, lane, tid, a0, a1, g, be, sLN, sMu, sRv, z);
#pragma unroll
    for (int v = 0; v < 8; ++v) {
      const float tm = half_max(z[v]);
      const float nm = fmaxf(m_run[v], tm);
      const float e  = __expf(z[v] - nm);
      const float es = half_sum(e);
      s_run[v] = s_run[v] * __expf(m_run[v] - nm) + es;
      m_run[v] = nm;
    }
  }
  float invs[8];
#pragma unroll
  for (int v = 0; v < 8; ++v) invs[v] = (s_run[v] > 0.f) ? (1.0f / s_run[v]) : 0.f;

  // ---- pass 2: P@V + column sums ----
  v8f O[4] = {{}, {}, {}, {}};
  _Float16* sPw = sP + head * (16 * 32);
  for (int j0 = 0; j0 < LL; j0 += 32) {
    // block-cooperative coalesced staging of V slab: rows j0..j0+31, 768 cols
#pragma unroll
    for (int it = 0; it < 16; ++it) {
      const int gq = tid + it * 384;          // 0 .. 6143 float4-groups
      const int jj = gq / 192, c4 = (gq % 192) * 4;
      const float4 x = *(const float4*)(hb + (size_t)(j0 + jj) * D3 + 2 * DD + c4);
      v4h h4; h4[0] = (_Float16)x.x; h4[1] = (_Float16)x.y;
              h4[2] = (_Float16)x.z; h4[3] = (_Float16)x.w;
      *(v4h*)(&sV[jj * DD + c4]) = h4;
    }
    float z0[8], z1[8];
    score_tile(hb, j0,      head, lane, tid, a0, a1, g, be, sLN, sMu, sRv, z0);
    score_tile(hb, j0 + 16, head, lane, tid, a0, a1, g, be, sLN, sMu, sRv, z1);
    float cs0 = 0.f, cs1 = 0.f;
#pragma unroll
    for (int v = 0; v < 8; ++v) {
      const float p0 = __expf(z0[v] - m_run[v]) * invs[v];  // removeNaN: invs==0
      const float p1 = __expf(z1[v] - m_run[v]) * invs[v];
      cs0 += p0; cs1 += p1;
      sPw[(mb + v) * 32 + n]      = (_Float16)p0;
      sPw[(mb + v) * 32 + 16 + n] = (_Float16)p1;
    }
    __syncthreads();
    // colsum over the 16 i-rows (mean over heads folded in)
    cs0 += __shfl_xor(cs0, 16);
    cs1 += __shfl_xor(cs1, 16);
    if (lane < 16) atomicAdd(&colAcc[j0 + n],      cs0 * (1.0f / HH));
    else           atomicAdd(&colAcc[j0 + 16 + n], cs1 * (1.0f / HH));
    // P as A-fragment (16 x 32)
    const v16h pa = load_frag_a(sPw, 32, lane);
    // V fragments from LDS: 32(j) x 64(d) -> 4 B-fragments
    const int kb = (lane >> 4) * 16;
#pragma unroll
    for (int c = 0; c < 4; ++c) {
      v16h vb;
      const int dcol = head * 64 + c * 16 + n;
#pragma unroll
      for (int t = 0; t < 16; ++t) vb[t] = sV[(kb + t) * DD + dcol];
      O[c] = wmma_f16(pa, vb, O[c]);
    }
    __syncthreads();
  }

  // epilogue: catHiddens layout obuf[b, i, head*64 + d]
#pragma unroll
  for (int c = 0; c < 4; ++c)
#pragma unroll
    for (int v = 0; v < 8; ++v)
      obuf[(size_t)(b * LL + i0 + mb + v) * DD + head * 64 + c * 16 + n] = O[c][v];

  __syncthreads();
  for (int i = tid; i < LL; i += 384)
    atomicAdd(&colsum[(size_t)b * LL + i], colAcc[i]);
}

// -------- get_weights: softmax over L of masks * colsum, per batch ---------
__global__ __launch_bounds__(512) void weights_kernel(
    const float* __restrict__ colsum, const int* __restrict__ ids,
    float* __restrict__ wbuf) {
  __shared__ float red[LL];
  const int b = blockIdx.x, l = threadIdx.x;
  const float v = (ids[(size_t)b * LL + l] != 0) ? colsum[(size_t)b * LL + l] : 0.f;
  red[l] = v;
  __syncthreads();
  for (int s = LL / 2; s > 0; s >>= 1) {
    if (l < s) red[l] = fmaxf(red[l], red[l + s]);
    __syncthreads();
  }
  const float mx = red[0];
  __syncthreads();
  const float e = expf(v - mx);
  red[l] = e;
  __syncthreads();
  for (int s = LL / 2; s > 0; s >>= 1) {
    if (l < s) red[l] += red[l + s];
    __syncthreads();
  }
  wbuf[(size_t)b * LL + l] = e / red[0];
}

// ------ out-projection + softmax(C=20) + weighted accumulate into d_out ----
__global__ __launch_bounds__(32) void outproj_kernel(
    const float* __restrict__ obuf, const float* __restrict__ Wout,
    const float* __restrict__ bout, const float* __restrict__ wbuf,
    float* __restrict__ dout) {
  const int row  = blockIdx.x;           // 0..8191 = b*L + l
  const int lane = threadIdx.x;
  float logit = -INFINITY;
  if (lane < CC) {
    logit = bout[lane];
    const float4* o4 = (const float4*)(obuf + (size_t)row * DD);
    const float4* w4 = (const float4*)(Wout + (size_t)lane * DD);
    float acc = 0.f;
    for (int k = 0; k < DD / 4; ++k) {
      const float4 a = o4[k], w = w4[k];
      acc += a.x * w.x + a.y * w.y + a.z * w.z + a.w * w.w;
    }
    logit += acc;
  }
  float mx = logit;
#pragma unroll
  for (int off = 16; off >= 1; off >>= 1) mx = fmaxf(mx, __shfl_xor(mx, off));
  float e = (lane < CC) ? expf(logit - mx) : 0.f;
  float sum = e;
#pragma unroll
  for (int off = 16; off >= 1; off >>= 1) sum += __shfl_xor(sum, off);
  if (lane < CC) {
    const int b = row >> 9;  // / LL
    atomicAdd(&dout[(size_t)b * CC + lane], 0.5f * wbuf[row] * (e / sum));
  }
}

// ---------------------------------------------------------------------------
extern "C" void kernel_launch(void* const* d_in, const int* in_sizes, int n_in,
                              void* d_out, int out_size, void* d_ws, size_t ws_size,
                              hipStream_t stream) {
  (void)in_sizes; (void)n_in; (void)out_size; (void)ws_size;
  const int*   ids     = (const int*)d_in[0];
  const float* wv      = (const float*)d_in[1];
  const float* emb     = (const float*)d_in[2];
  const float* W_t     = (const float*)d_in[3];
  const float* b_t     = (const float*)d_in[4];
  const float* gamma_t = (const float*)d_in[5];
  const float* beta_t  = (const float*)d_in[6];
  const float* Wout_t  = (const float*)d_in[7];
  const float* bout_t  = (const float*)d_in[8];
  const float* W_h     = (const float*)d_in[9];
  const float* b_h     = (const float*)d_in[10];
  const float* gamma_h = (const float*)d_in[11];
  const float* beta_h  = (const float*)d_in[12];
  const float* Wout_h  = (const float*)d_in[13];
  const float* bout_h  = (const float*)d_in[14];
  float* out = (float*)d_out;

  const size_t M = (size_t)BB * LL;  // 8192
  char* ws = (char*)d_ws;
  float* xg     = (float*)ws; ws += M * DD * sizeof(float);  // also wvp [M,KH]
  float* hbuf   = (float*)ws; ws += M * D3 * sizeof(float);
  float* obuf   = (float*)ws; ws += M * DD * sizeof(float);
  float* colsum = (float*)ws; ws += M * sizeof(float);
  float* wbuf   = (float*)ws; ws += M * sizeof(float);

  hipMemsetAsync(out, 0, (size_t)BB * CC * sizeof(float), stream);

  const dim3 gGemm(D3 / 32, (unsigned)(M / 64));        // (72, 128)
  const dim3 gAttn(LL / 16, BB);                        // (32, 16)
  const unsigned gGather = (unsigned)((M * DD / 4 + 255) / 256);
  const unsigned gPad    = (unsigned)((M * KH / 4 + 255) / 256);

  // ---------------- token branch ----------------
  gather_leaky_kernel<<<gGather, 256, 0, stream>>>(ids, emb, xg);
  hipMemsetAsync(colsum, 0, M * sizeof(float), stream);
  gemm_bias_leaky_kernel<<<gGemm, 256, 0, stream>>>(xg, W_t, b_t, hbuf,
                                                    (int)M, D3, DD);
  etc_attn_kernel<<<gAttn, 384, 0, stream>>>(hbuf, gamma_t, beta_t, obuf, colsum);
  weights_kernel<<<BB, LL, 0, stream>>>(colsum, ids, wbuf);
  outproj_kernel<<<(unsigned)M, 32, 0, stream>>>(obuf, Wout_t, bout_t, wbuf, out);

  // ---------------- hidden branch ----------------
  // Pad W_h's K dim on the fly? No: pad X (wv) to K=320; W_h rows are only 300
  // floats, so stage the weight through the same padded layout trick is not
  // needed -- instead pad wv AND read W_h with the K-tail handled by padding
  // the X operand only. Both operands must agree on K, so we pad W_h too.
  float* wvp = xg;               // reuse (token branch done with xg by now)
  pad_wv_kernel<<<gPad, 256, 0, stream>>>(wv, wvp);
  hipMemsetAsync(colsum, 0, M * sizeof(float), stream);
  // W_h padded into obuf head (2304*320 floats = 2.8 MB << obuf size)
  {
    float* whp = obuf;  // scratch before attention overwrites it
    const unsigned gPadW = (unsigned)(((size_t)D3 * KH / 4 + 255) / 256);
    // reuse pad kernel shape: rows = D3, cols 300 -> 320
    pad_wv_kernel<<<gPadW, 256, 0, stream>>>(W_h, whp);
    gemm_bias_leaky_kernel<<<gGemm, 256, 0, stream>>>(wvp, whp, b_h, hbuf,
                                                      (int)M, D3, KH);
  }
  etc_attn_kernel<<<gAttn, 384, 0, stream>>>(hbuf, gamma_h, beta_h, obuf, colsum);
  weights_kernel<<<BB, LL, 0, stream>>>(colsum, ids, wbuf);
  outproj_kernel<<<(unsigned)M, 32, 0, stream>>>(obuf, Wout_h, bout_h, wbuf, out);
}
